// Encoder_36730560315530
// MI455X (gfx1250) — compile-verified, matrix operand order checked by reading, not run on hardware
//
#include <hip/hip_runtime.h>
#include <cstdint>
#include <cstddef>

// ---------------------------------------------------------------------------
// MI455X (gfx1250) encoder forward.
// All heavy math (QKV proj, QK^T, PV, Wo, FFN) runs through
// v_wmma_f32_16x16x32_bf16 (wave32, f32 accumulate).
// Workspace requirement: ~176 MB (see carve() below).
// ---------------------------------------------------------------------------

typedef __bf16 bf16_t;
typedef __attribute__((ext_vector_type(16))) __bf16 v16bf;
typedef __attribute__((ext_vector_type(8)))  __bf16 v8bf;
typedef __attribute__((ext_vector_type(8)))  float  v8f;

#define TILE_M 128
#define TILE_N 128
#define TILE_K 32
#define LDSK   40   // 32 + 8 bf16 pad -> 80B row stride, keeps 16B alignment

__device__ __forceinline__ float wave_max(float v) {
#pragma unroll
  for (int o = 16; o > 0; o >>= 1) v = fmaxf(v, __shfl_xor(v, o, 32));
  return v;
}
__device__ __forceinline__ float wave_sum(float v) {
#pragma unroll
  for (int o = 16; o > 0; o >>= 1) v += __shfl_xor(v, o, 32);
  return v;
}

// ---------------------------------------------------------------------------
// Batched bf16 WMMA GEMM:  C[z] = epilogue( A[z] (MxK) * B[z] (KxN) )
//   z = o*H + h ; per-operand two-level batch strides (elements).
//   BT:       B stored as [N x K] row-major (used for Q*K^T).
//   OUT_BF16: write bf16 (feeds the next WMMA GEMM) else f32.
//   epilogue: v = acc*scale + bias[n] ; optional ReLU.
// Block: 256 thr = 8 wave32, tile 128x128x32; wave owns 32x64 (2x4 frags).
// ---------------------------------------------------------------------------
template <bool BT, bool OUT_BF16, bool RELU>
__global__ __launch_bounds__(256) void wmma_gemm_bf16(
    const bf16_t* __restrict__ Ag, long long lda, long long sAo, long long sAh,
    const bf16_t* __restrict__ Bg, long long ldb, long long sBo, long long sBh,
    void* __restrict__ Cg, long long ldc, long long sCo, long long sCh,
    const float* __restrict__ bias, float scale,
    int M, int N, int K, int H)
{
  __shared__ bf16_t As[TILE_M][LDSK];
  __shared__ bf16_t Bs[TILE_N][LDSK];   // stored K-contiguous: Bs[n][k]

  const int z = blockIdx.z, o = z / H, h = z - o * H;
  const bf16_t* A = Ag + (long long)o * sAo + (long long)h * sAh;
  const bf16_t* B = Bg + (long long)o * sBo + (long long)h * sBh;
  const long long coff = (long long)o * sCo + (long long)h * sCh;

  const int bm = blockIdx.y * TILE_M;
  const int bn = blockIdx.x * TILE_N;
  const int tid  = threadIdx.x;
  const int wave = tid >> 5;
  const int lane = tid & 31;
  const int wm = (wave & 3) * 32;     // 4 waves along M
  const int wn = (wave >> 2) * 64;    // 2 waves along N
  const int lrow = lane & 15;
  const int kb   = (lane >> 4) << 3;  // K-base of this lane's fragment half

  v8f acc[2][4] = {};

  for (int k0 = 0; k0 < K; k0 += TILE_K) {
    // ---- stage A tile (128x32), 8B vector loads, contiguous stores ----
#pragma unroll
    for (int it = 0; it < 4; ++it) {
      int idx = tid + it * 256;
      int r = idx >> 3;
      int c = (idx & 7) << 2;
      *(uint64_t*)&As[r][c] =
          *(const uint64_t*)&A[(long long)(bm + r) * lda + (k0 + c)];
    }
    if (BT) {
      // B is [N x K]: rows are already K-contiguous
#pragma unroll
      for (int it = 0; it < 4; ++it) {
        int idx = tid + it * 256;
        int nn = idx >> 3;
        int c = (idx & 7) << 2;
        *(uint64_t*)&Bs[nn][c] =
            *(const uint64_t*)&B[(long long)(bn + nn) * ldb + (k0 + c)];
      }
    } else {
      // B is [K x N]: transpose on LDS store so fragment loads stay b128
#pragma unroll
      for (int it = 0; it < 4; ++it) {
        int idx = tid + it * 256;
        int kk = idx >> 5;
        int nn = (idx & 31) << 2;
        uint64_t pk = *(const uint64_t*)&B[(long long)(k0 + kk) * ldb + (bn + nn)];
        const bf16_t* pe = (const bf16_t*)&pk;
        Bs[nn + 0][kk] = pe[0];
        Bs[nn + 1][kk] = pe[1];
        Bs[nn + 2][kk] = pe[2];
        Bs[nn + 3][kk] = pe[3];
      }
    }
    if (k0 + TILE_K < K)   // hint next K-tile into cache (global_prefetch_b8)
      __builtin_prefetch(&A[(long long)(bm + (tid >> 1)) * lda + (k0 + TILE_K)], 0, 1);
    __syncthreads();

    // ---- fragments: lane holds row lrow, K = kb..kb+7 and kb+16..kb+23 ----
    v16bf afrag[2], bfrag[4];
#pragma unroll
    for (int i = 0; i < 2; ++i) {
      const bf16_t* p = &As[wm + i * 16 + lrow][kb];
      v8bf lo = *(const v8bf*)p;
      v8bf hi = *(const v8bf*)(p + 16);
#pragma unroll
      for (int e = 0; e < 8; ++e) { afrag[i][e] = lo[e]; afrag[i][e + 8] = hi[e]; }
    }
#pragma unroll
    for (int j = 0; j < 4; ++j) {
      const bf16_t* p = &Bs[wn + j * 16 + lrow][kb];
      v8bf lo = *(const v8bf*)p;
      v8bf hi = *(const v8bf*)(p + 16);
#pragma unroll
      for (int e = 0; e < 8; ++e) { bfrag[j][e] = lo[e]; bfrag[j][e + 8] = hi[e]; }
    }
#pragma unroll
    for (int i = 0; i < 2; ++i)
#pragma unroll
      for (int j = 0; j < 4; ++j)
        acc[i][j] = __builtin_amdgcn_wmma_f32_16x16x32_bf16(
            false, afrag[i], false, bfrag[j], (short)0, acc[i][j], false, false);
    __syncthreads();
  }

  // ---- epilogue: C-frag layout VGPR r -> M = r + 8*(lane>=16), N = lane%16 --
  const int mlo = (lane >> 4) * 8;
#pragma unroll
  for (int i = 0; i < 2; ++i) {
#pragma unroll
    for (int j = 0; j < 4; ++j) {
      int gcol = bn + wn + j * 16 + lrow;
      float bv = bias ? bias[gcol] : 0.0f;
#pragma unroll
      for (int r = 0; r < 8; ++r) {
        int grow = bm + wm + i * 16 + mlo + r;
        float v = acc[i][j][r] * scale + bv;
        if (RELU) v = fmaxf(v, 0.0f);
        long long ci = coff + (long long)grow * ldc + gcol;
        if (OUT_BF16) ((bf16_t*)Cg)[ci] = (bf16_t)v;
        else          ((float*)Cg)[ci]  = v;
      }
    }
  }
}

// ---------------------------------------------------------------------------
// Row softmax over fp32 scores -> bf16 probabilities. One block per row.
// (1/sqrt(KD) scale already folded into the score GEMM.)
// ---------------------------------------------------------------------------
__global__ __launch_bounds__(256) void softmax_bf16(
    const float* __restrict__ S, bf16_t* __restrict__ P, int N)
{
  __shared__ float red[8];
  const long long row = blockIdx.x;
  const float* x = S + row * (long long)N;
  bf16_t* p = P + row * (long long)N;
  const int t = threadIdx.x;

  float m = -3.4e38f;
  for (int i = t; i < N; i += 256) m = fmaxf(m, x[i]);
  m = wave_max(m);
  if ((t & 31) == 0) red[t >> 5] = m;
  __syncthreads();
  float rm = red[0];
#pragma unroll
  for (int i = 1; i < 8; ++i) rm = fmaxf(rm, red[i]);
  __syncthreads();

  float s = 0.0f;
  for (int i = t; i < N; i += 256) s += __expf(x[i] - rm);
  s = wave_sum(s);
  if ((t & 31) == 0) red[t >> 5] = s;
  __syncthreads();
  float rs = 0.0f;
#pragma unroll
  for (int i = 0; i < 8; ++i) rs += red[i];
  const float inv = 1.0f / rs;
  for (int i = t; i < N; i += 256) p[i] = (bf16_t)(__expf(x[i] - rm) * inv);
}

// ---------------------------------------------------------------------------
// out = LayerNorm(a + b) * g + beta  (D = 512, eps = 1e-3). Also emits a bf16
// copy (for feeding the next WMMA GEMM) when outb != nullptr.
// ---------------------------------------------------------------------------
__global__ __launch_bounds__(256) void add_layernorm512(
    const float* __restrict__ A, const float* __restrict__ Br,
    const float* __restrict__ g, const float* __restrict__ be,
    float* __restrict__ outf, bf16_t* __restrict__ outb)
{
  __shared__ float red[8];
  const long long base = (long long)blockIdx.x * 512;
  const int t = threadIdx.x;
  float v0 = A[base + t]       + Br[base + t];
  float v1 = A[base + t + 256] + Br[base + t + 256];

  float s = wave_sum(v0 + v1);
  if ((t & 31) == 0) red[t >> 5] = s;
  __syncthreads();
  float tot = 0.0f;
#pragma unroll
  for (int i = 0; i < 8; ++i) tot += red[i];
  const float mu = tot * (1.0f / 512.0f);
  __syncthreads();

  float d0 = v0 - mu, d1 = v1 - mu;
  float vs = wave_sum(d0 * d0 + d1 * d1);
  if ((t & 31) == 0) red[t >> 5] = vs;
  __syncthreads();
  float var = 0.0f;
#pragma unroll
  for (int i = 0; i < 8; ++i) var += red[i];
  var *= (1.0f / 512.0f);
  const float rsi = rsqrtf(var + 1e-3f);

  float y0 = d0 * rsi * g[t]       + be[t];
  float y1 = d1 * rsi * g[t + 256] + be[t + 256];
  outf[base + t]       = y0;
  outf[base + t + 256] = y1;
  if (outb) {
    outb[base + t]       = (bf16_t)y0;
    outb[base + t + 256] = (bf16_t)y1;
  }
}

__global__ void cvt_f32_bf16(const float* __restrict__ in,
                             bf16_t* __restrict__ out, long long n)
{
  long long i  = (long long)blockIdx.x * blockDim.x + threadIdx.x;
  long long st = (long long)gridDim.x * blockDim.x;
  for (; i < n; i += st) out[i] = (bf16_t)in[i];
}

__global__ void copy_cast_f32(const float* __restrict__ in,
                              float* __restrict__ outf,
                              bf16_t* __restrict__ outb, long long n)
{
  long long i  = (long long)blockIdx.x * blockDim.x + threadIdx.x;
  long long st = (long long)gridDim.x * blockDim.x;
  for (; i < n; i += st) { float v = in[i]; outf[i] = v; outb[i] = (bf16_t)v; }
}

// ---------------------------------------------------------------------------
// Host-side orchestration
// ---------------------------------------------------------------------------
namespace {
struct Mha  { const float *Wk, *Wo, *Wq, *Wv, *bk, *bo, *bq, *bv; };
struct Attn { const float *ln_b, *ln_g; Mha m; };
struct Ffn  { const float *W1, *W2, *b1, *b2, *ln_b, *ln_g; };
struct Layer { Attn gsa; Ffn gsa_ffn; Attn lsa; Ffn lsa_ffn; };
} // namespace

extern "C" void kernel_launch(void* const* d_in, const int* in_sizes, int n_in,
                              void* d_out, int out_size, void* d_ws, size_t ws_size,
                              hipStream_t stream)
{
  (void)in_sizes;
  constexpr int Lc = 2, Dm = 512, HG = 8, HL = 2, Wn = 8, DFF = 2048, KD = 512;
  constexpr int Bb = 4, Ss = 1024, NT = Bb * Ss;   // 4096 tokens
  if (n_in < 2 * 32 + 3) return;

  // Assumed flatten order: jax pytree (dict keys sorted, lists in order):
  // per layer: gsa{ln_b,ln_g,Wk,Wo,Wq,Wv,bk,bo,bq,bv}, gsa_ffn{W1,W2,b1,b2,ln_b,ln_g},
  //            lsa{...}, lsa_ffn{...}; then final ln_b, ln_g; then x.
  int idx = 0;
  auto nx = [&]() -> const float* { return (const float*)d_in[idx++]; };
  Layer ly[Lc];
  for (int l = 0; l < Lc; ++l) {
    Attn& g = ly[l].gsa;
    g.ln_b = nx(); g.ln_g = nx();
    g.m.Wk = nx(); g.m.Wo = nx(); g.m.Wq = nx(); g.m.Wv = nx();
    g.m.bk = nx(); g.m.bo = nx(); g.m.bq = nx(); g.m.bv = nx();
    Ffn& gf = ly[l].gsa_ffn;
    gf.W1 = nx(); gf.W2 = nx(); gf.b1 = nx(); gf.b2 = nx(); gf.ln_b = nx(); gf.ln_g = nx();
    Attn& s = ly[l].lsa;
    s.ln_b = nx(); s.ln_g = nx();
    s.m.Wk = nx(); s.m.Wo = nx(); s.m.Wq = nx(); s.m.Wv = nx();
    s.m.bk = nx(); s.m.bo = nx(); s.m.bq = nx(); s.m.bv = nx();
    Ffn& sf = ly[l].lsa_ffn;
    sf.W1 = nx(); sf.W2 = nx(); sf.b1 = nx(); sf.b2 = nx(); sf.ln_b = nx(); sf.ln_g = nx();
  }
  const float* fin_ln_b = nx();
  const float* fin_ln_g = nx();
  const float* x_in     = nx();

  // ---- workspace carve (~176 MB, 256B aligned) ----
  char* wsp = (char*)d_ws;
  size_t off = 0;
  auto carve = [&](size_t bytes) -> void* {
    void* p = wsp + off;
    off = (off + bytes + 255) & ~(size_t)255;
    return p;
  };
  float*  x_cur  = (float*) carve((size_t)NT * Dm * 4);
  bf16_t* xbf    = (bf16_t*)carve((size_t)NT * Dm * 2);
  bf16_t* qbf    = (bf16_t*)carve((size_t)1024 * 4096 * 2);  // per-b GSA / all-b LSA
  bf16_t* kbf    = (bf16_t*)carve((size_t)1024 * 4096 * 2);
  bf16_t* vbf    = (bf16_t*)carve((size_t)1024 * 4096 * 2);
  float*  scores = (float*) carve((size_t)HG * Ss * Ss * 4); // 32 MB, reused
  bf16_t* probs  = (bf16_t*)carve((size_t)HG * Ss * Ss * 2); // 16 MB
  bf16_t* obuf   = (bf16_t*)carve((size_t)1024 * 4096 * 2);
  float*  ao     = (float*) carve((size_t)NT * Dm * 4);
  float*  tact   = (float*) carve((size_t)NT * Dm * 4);
  bf16_t* tbf    = (bf16_t*)carve((size_t)NT * Dm * 2);
  bf16_t* h1     = (bf16_t*)carve((size_t)NT * DFF * 2);
  float*  yf     = (float*) carve((size_t)NT * Dm * 4);
  float*  lsaout = (float*) carve((size_t)NT * Dm * 4);
  float*  gsaout = (float*) carve((size_t)NT * Dm * 4);
  bf16_t* wslot[4];
  for (int i = 0; i < 4; ++i) wslot[i] = (bf16_t*)carve((size_t)512 * 4096 * 2);
  (void)ws_size;

  auto cvt = [&](const float* src, bf16_t* dst, long long n) {
    int blocks = (int)((n + 1023) / 1024);
    cvt_f32_bf16<<<dim3(blocks), dim3(256), 0, stream>>>(src, dst, n);
  };

  auto gemm = [&](bool bt, bool obf, bool relu,
                  const bf16_t* A, long long lda, long long sAo, long long sAh,
                  const bf16_t* B, long long ldb, long long sBo, long long sBh,
                  void* C, long long ldc, long long sCo, long long sCh,
                  const float* bias, float scale, int M, int N, int K, int O, int H) {
    dim3 grid(N / TILE_N, M / TILE_M, O * H), block(256);
    if (bt)
      wmma_gemm_bf16<true, false, false><<<grid, block, 0, stream>>>(
          A, lda, sAo, sAh, B, ldb, sBo, sBh, C, ldc, sCo, sCh, bias, scale, M, N, K, H);
    else if (obf && relu)
      wmma_gemm_bf16<false, true, true><<<grid, block, 0, stream>>>(
          A, lda, sAo, sAh, B, ldb, sBo, sBh, C, ldc, sCo, sCh, bias, scale, M, N, K, H);
    else if (obf)
      wmma_gemm_bf16<false, true, false><<<grid, block, 0, stream>>>(
          A, lda, sAo, sAh, B, ldb, sBo, sBh, C, ldc, sCo, sCh, bias, scale, M, N, K, H);
    else
      wmma_gemm_bf16<false, false, false><<<grid, block, 0, stream>>>(
          A, lda, sAo, sAh, B, ldb, sBo, sBh, C, ldc, sCo, sCh, bias, scale, M, N, K, H);
  };

  const float inv_sqrt_kd = 0.044194173824159216f;  // 1/sqrt(512)

  auto run_ffn = [&](const Ffn& f, const float* resid, const bf16_t* rbf, float* out) {
    cvt(f.W1, wslot[0], (long long)Dm * DFF);
    cvt(f.W2, wslot[1], (long long)DFF * Dm);
    gemm(false, true, true,  rbf, Dm, 0, 0, wslot[0], DFF, 0, 0,
         h1, DFF, 0, 0, f.b1, 1.0f, NT, DFF, Dm, 1, 1);                 // relu(x@W1+b1)
    gemm(false, false, false, h1, DFF, 0, 0, wslot[1], Dm, 0, 0,
         yf, Dm, 0, 0, f.b2, 1.0f, NT, Dm, DFF, 1, 1);                  // h@W2+b2
    add_layernorm512<<<NT, 256, 0, stream>>>(resid, yf, f.ln_g, f.ln_b, out, nullptr);
  };

  auto run_lsa = [&](const Attn& a, const Ffn& f, float* bout) {
    const int Np = HL * KD;          // 1024
    const int Sw = Ss / Wn;          // 128 tokens per window
    cvt(a.m.Wq, wslot[0], (long long)Dm * Np);
    cvt(a.m.Wk, wslot[1], (long long)Dm * Np);
    cvt(a.m.Wv, wslot[2], (long long)Dm * Np);
    cvt(a.m.Wo, wslot[3], (long long)Np * Dm);
    gemm(false, true, false, xbf, Dm, 0, 0, wslot[0], Np, 0, 0,
         qbf, Np, 0, 0, a.m.bq, 1.0f, NT, Np, Dm, 1, 1);
    gemm(false, true, false, xbf, Dm, 0, 0, wslot[1], Np, 0, 0,
         kbf, Np, 0, 0, a.m.bk, 1.0f, NT, Np, Dm, 1, 1);
    gemm(false, true, false, xbf, Dm, 0, 0, wslot[2], Np, 0, 0,
         vbf, Np, 0, 0, a.m.bv, 1.0f, NT, Np, Dm, 1, 1);
    const long long so = (long long)Sw * Np;       // per-window q/k/v stride
    // scores[b*W+w, h] = (Q K^T) / sqrt(KD)  (batched: O = B*W, H = HL)
    gemm(true, false, false, qbf, Np, so, KD, kbf, Np, so, KD,
         scores, Sw, (long long)HL * Sw * Sw, (long long)Sw * Sw,
         nullptr, inv_sqrt_kd, Sw, Sw, KD, Bb * Wn, HL);
    softmax_bf16<<<Bb * Wn * HL * Sw, 256, 0, stream>>>(scores, probs, Sw);
    gemm(false, true, false, probs, Sw, (long long)HL * Sw * Sw, (long long)Sw * Sw,
         vbf, Np, so, KD, obuf, Np, so, KD, nullptr, 1.0f, Sw, KD, Sw, Bb * Wn, HL);
    gemm(false, false, false, obuf, Np, 0, 0, wslot[3], Dm, 0, 0,
         ao, Dm, 0, 0, a.m.bo, 1.0f, NT, Dm, Np, 1, 1);
    add_layernorm512<<<NT, 256, 0, stream>>>(x_cur, ao, a.ln_g, a.ln_b, tact, tbf);
    run_ffn(f, tact, tbf, bout);
  };

  auto run_gsa = [&](const Attn& a, const Ffn& f, float* bout) {
    const int Np = HG * KD;          // 4096
    cvt(a.m.Wq, wslot[0], (long long)Dm * Np);
    cvt(a.m.Wk, wslot[1], (long long)Dm * Np);
    cvt(a.m.Wv, wslot[2], (long long)Dm * Np);
    cvt(a.m.Wo, wslot[3], (long long)Np * Dm);
    for (int b = 0; b < Bb; ++b) {   // chunk over batch to bound workspace
      const bf16_t* xb = xbf + (long long)b * Ss * Dm;
      gemm(false, true, false, xb, Dm, 0, 0, wslot[0], Np, 0, 0,
           qbf, Np, 0, 0, a.m.bq, 1.0f, Ss, Np, Dm, 1, 1);
      gemm(false, true, false, xb, Dm, 0, 0, wslot[1], Np, 0, 0,
           kbf, Np, 0, 0, a.m.bk, 1.0f, Ss, Np, Dm, 1, 1);
      gemm(false, true, false, xb, Dm, 0, 0, wslot[2], Np, 0, 0,
           vbf, Np, 0, 0, a.m.bv, 1.0f, Ss, Np, Dm, 1, 1);
      gemm(true, false, false, qbf, Np, 0, KD, kbf, Np, 0, KD,
           scores, Ss, 0, (long long)Ss * Ss, nullptr, inv_sqrt_kd,
           Ss, Ss, KD, 1, HG);
      softmax_bf16<<<HG * Ss, 256, 0, stream>>>(scores, probs, Ss);
      gemm(false, true, false, probs, Ss, 0, (long long)Ss * Ss,
           vbf, Np, 0, KD, obuf, Np, 0, KD, nullptr, 1.0f, Ss, KD, Ss, 1, HG);
      gemm(false, false, false, obuf, Np, 0, 0, wslot[3], Dm, 0, 0,
           ao + (long long)b * Ss * Dm, Dm, 0, 0, a.m.bo, 1.0f, Ss, Dm, Np, 1, 1);
    }
    add_layernorm512<<<NT, 256, 0, stream>>>(x_cur, ao, a.ln_g, a.ln_b, tact, tbf);
    run_ffn(f, tact, tbf, bout);
  };

  // ---- encoder forward ----
  copy_cast_f32<<<2048, 256, 0, stream>>>(x_in, x_cur, xbf, (long long)NT * Dm);
  for (int l = 0; l < Lc; ++l) {
    run_lsa(ly[l].lsa, ly[l].lsa_ffn, lsaout);   // both branches read the same x
    run_gsa(ly[l].gsa, ly[l].gsa_ffn, gsaout);
    add_layernorm512<<<NT, 256, 0, stream>>>(lsaout, gsaout, fin_ln_g, fin_ln_b,
                                             x_cur, xbf);
  }
  hipMemcpyAsync(d_out, x_cur, (size_t)out_size * sizeof(float),
                 hipMemcpyDeviceToDevice, stream);
}
